// BiLevelRoutingAttention_53996328845633
// MI455X (gfx1250) — matrix operand order repaired and checked by reading, not compiled, not run on hardware
//
#include <hip/hip_runtime.h>
#include <math.h>

// ---------------- CDNA5 WMMA types ----------------
typedef __attribute__((ext_vector_type(16))) _Float16 v16h;
typedef __attribute__((ext_vector_type(8)))  _Float16 v8h;
typedef __attribute__((ext_vector_type(8)))  float    v8f;

#define T_DIM 4
#define B_DIM 4
#define L_DIM 1024
#define C_DIM 256
#define NWIN  8
#define WIN   128
#define NH    8
#define HD    32
#define TOPK  4

static __device__ __forceinline__ v8f wmma16x16x32(v16h a, v16h b, v8f c) {
  // D = A(16x32 f16) * B(32x16 f16) + C(16x16 f32)
  return __builtin_amdgcn_wmma_f32_16x16x32_f16(
      /*neg_a=*/false, a, /*neg_b=*/false, b,
      /*c_mod=*/(short)0, c, /*reuse_a=*/false, /*reuse_b=*/false);
}

// Fragment index maps (ISA 7.12.2, wave32):
//  A 16x32 f16 : row = lane%16,          K(e) = (e%8) + 16*(e/8) + 8*(lane/16)
//  B 32x16 f16 : col = lane%16,          K(e) = (e%8) + 16*(e/8) + 8*(lane/16)
//  C/D 16x16 f32: row = r + 8*(lane/16), col = lane%16

// ---------------- Stage 1a: region sums ----------------
__global__ __launch_bounds__(256)
void region_sum_kernel(const float* __restrict__ x, float* __restrict__ region) {
  const int bn = blockIdx.x;            // b*8+n
  const int b = bn >> 3, n = bn & 7;
  const int c = threadIdx.x;
  float s = 0.f;
  for (int t = 0; t < T_DIM; ++t) {
    const float* base = x + ((size_t)(t * B_DIM + b) * L_DIM + n * WIN) * C_DIM + c;
    for (int r = 0; r < WIN; ++r) s += base[(size_t)r * C_DIM];
  }
  region[(size_t)bn * C_DIM + c] = s;
}

// ---------------- Stage 1b: 8x8 gram + stable top-4 ----------------
__global__ __launch_bounds__(64)
void route_topk_kernel(const float* __restrict__ region, int* __restrict__ idx) {
  __shared__ float reg[NWIN][C_DIM];
  __shared__ float ar[NWIN][NWIN];
  const int b = blockIdx.x;
  for (int i = threadIdx.x; i < NWIN * C_DIM; i += 64)
    reg[i / C_DIM][i % C_DIM] = region[(size_t)b * NWIN * C_DIM + i];
  __syncthreads();
  {
    const int n = threadIdx.x >> 3, m = threadIdx.x & 7;
    float d = 0.f;
    for (int c = 0; c < C_DIM; ++c) d += reg[n][c] * reg[m][c];
    ar[n][m] = d * 0.0625f;             // * C^-0.5
  }
  __syncthreads();
  if (threadIdx.x < NWIN) {
    const int n = threadIdx.x;
    float vals[NWIN];
    for (int m = 0; m < NWIN; ++m) vals[m] = ar[n][m];
    for (int kk = 0; kk < TOPK; ++kk) {   // stable: strict > keeps lowest idx on tie
      int bi = 0; float bvv = vals[0];
      for (int m = 1; m < NWIN; ++m) if (vals[m] > bvv) { bvv = vals[m]; bi = m; }
      idx[((size_t)b * NWIN + n) * TOPK + kk] = bi;
      vals[bi] = -INFINITY;
    }
  }
}

// ---------------- Weight pack: [K][N] f32 -> B-fragment-ordered f16 ----------------
// wsw[((nt*(K/32)+kc)*32 + lane)*16 + e] = W[kc*32 + (e%8) + 16*(e/8) + 8*(lane/16)][nt*16 + lane%16]
__global__ __launch_bounds__(256)
void pack_w_kernel(const float* __restrict__ W, _Float16* __restrict__ wsw,
                   int K, int N) {
  const size_t gid = (size_t)blockIdx.x * 256 + threadIdx.x;
  if (gid >= (size_t)K * N) return;
  const int e = (int)(gid & 15);
  const int lane = (int)((gid >> 4) & 31);
  const size_t tile = gid >> 9;          // nt*(K/32)+kc
  const int KC = K >> 5;
  const int kc = (int)(tile % KC);
  const int nt = (int)(tile / KC);
  const int k = kc * 32 + (e & 7) + 16 * (e >> 3) + 8 * (lane >> 4);
  const int n = nt * 16 + (lane & 15);
  wsw[gid] = (_Float16)W[(size_t)k * N + n];
}

// ---------------- WMMA GEMM: C = A*Bsw + bias ----------------
// K in 64-wide quarters: A quarter-panel (128x64 f32->f16) + B fragments
// (2kc x 8tc, fragment-ordered bulk copy) staged in LDS behind one barrier
// pair. Inner loop: statically-unrolled 16-WMMA chain with a 1-deep
// software pipeline on B fragments (b_cur/b_next) so DS latency hides
// behind the previous WMMA.
__global__ __launch_bounds__(256)
void gemm_f16_wmma(const float* __restrict__ A, const _Float16* __restrict__ Bsw,
                   const float* __restrict__ bias, float* __restrict__ C,
                   int M, int N, int K) {
  // A pitch 72 halfs = 144B: 16B-aligned frag runs; 36*lr mod 64 -> distinct banks
  __shared__ __align__(16) _Float16 As[128][72];       // 18 KB
  __shared__ __align__(16) _Float16 Bl[2 * 8 * 512];   // 16 KB
  const int tid = threadIdx.x;
  const int wave = tid >> 5, lane = tid & 31;
  const int lg = lane >> 4, lr = lane & 15;
  const int mblk = blockIdx.y, nblk = blockIdx.x;
  const int KC = K >> 5;
  const int NQ = K >> 6;

  v8f acc[8];
  v8f zero = {};
  #pragma unroll
  for (int i = 0; i < 8; ++i) acc[i] = zero;

  for (int q = 0; q < NQ; ++q) {
    __syncthreads();   // previous quarter's LDS reads complete
    // stage A 128x64 quarter: vector f32 loads -> f16 -> vector LDS stores
    #pragma unroll
    for (int it = 0; it < 2; ++it) {
      const int r = tid >> 1;
      const int cb = (tid & 1) * 16 + it * 32;
      const float4* src =
          (const float4*)(A + (size_t)(mblk * 128 + r) * K + q * 64 + cb);
      const float4 f0 = src[0], f1 = src[1], f2 = src[2], f3 = src[3];
      v8h h0, h1;
      h0[0]=(_Float16)f0.x; h0[1]=(_Float16)f0.y; h0[2]=(_Float16)f0.z; h0[3]=(_Float16)f0.w;
      h0[4]=(_Float16)f1.x; h0[5]=(_Float16)f1.y; h0[6]=(_Float16)f1.z; h0[7]=(_Float16)f1.w;
      h1[0]=(_Float16)f2.x; h1[1]=(_Float16)f2.y; h1[2]=(_Float16)f2.z; h1[3]=(_Float16)f2.w;
      h1[4]=(_Float16)f3.x; h1[5]=(_Float16)f3.y; h1[6]=(_Float16)f3.z; h1[7]=(_Float16)f3.w;
      *(v8h*)&As[r][cb]     = h0;
      *(v8h*)&As[r][cb + 8] = h1;
    }
    // stage B fragments (2 kc x 8 tc x 512 halfs): bulk 16B copies
    #pragma unroll
    for (int it = 0; it < 4; ++it) {
      const int flat = it * 2048 + tid * 8;   // half index, 16B-aligned
      const int kc2 = flat >> 12;
      const int rem = flat & 4095;
      const int tc = rem >> 9;
      const int inner = rem & 511;
      const v8h d = *(const v8h*)(
          Bsw + ((size_t)(nblk * 8 + tc) * KC + q * 2 + kc2) * 512 + inner);
      *(v8h*)&Bl[flat] = d;
    }
    if (q + 1 < NQ)   // global_prefetch_b8 next A quarter
      __builtin_prefetch(A + (size_t)(mblk * 128 + (tid >> 1)) * K + q * 64 + 64, 0, 0);
    __syncthreads();

    // preload both A fragments of this quarter (2x ds_load_b128 each)
    const _Float16* ap0 = &As[wave * 16 + lr][8 * lg];
    const v8h a0lo = *(const v8h*)ap0;
    const v8h a0hi = *(const v8h*)(ap0 + 16);
    const v16h a0 = __builtin_shufflevector(a0lo, a0hi,
        0,1,2,3,4,5,6,7,8,9,10,11,12,13,14,15);
    const _Float16* ap1 = ap0 + 32;
    const v8h a1lo = *(const v8h*)ap1;
    const v8h a1hi = *(const v8h*)(ap1 + 16);
    const v16h a1 = __builtin_shufflevector(a1lo, a1hi,
        0,1,2,3,4,5,6,7,8,9,10,11,12,13,14,15);

    // 16-WMMA chain, B pipelined one step ahead
    v16h bcur = *(const v16h*)&Bl[lane * 16];   // s=0
    #pragma unroll
    for (int kc2 = 0; kc2 < 2; ++kc2) {
      const v16h a = kc2 ? a1 : a0;
      #pragma unroll
      for (int tc = 0; tc < 8; ++tc) {
        const int s = kc2 * 8 + tc;
        v16h bnext = bcur;
        if (s < 15)
          bnext = *(const v16h*)&Bl[((s + 1) * 512) + lane * 16];
        acc[tc] = wmma16x16x32(a, bcur, acc[tc]);
        bcur = bnext;
      }
    }
  }
  #pragma unroll
  for (int tc = 0; tc < 8; ++tc) {
    const int col = nblk * 128 + tc * 16 + lr;
    const float bv = bias ? bias[col] : 0.f;
    #pragma unroll
    for (int r = 0; r < 8; ++r) {
      const int row = mblk * 128 + wave * 16 + r + 8 * lg;
      C[(size_t)row * N + col] = acc[tc][r] + bv;
    }
  }
}

// ---------------- LIF over T: qkv -> fragment-ordered spike tensors ----------------
__global__ __launch_bounds__(256)
void lif_qkv_kernel(const float* __restrict__ qkv, _Float16* __restrict__ qs,
                    _Float16* __restrict__ ks, _Float16* __restrict__ vs) {
  const size_t gid = (size_t)blockIdx.x * 256 + threadIdx.x;  // over B*L*768
  const int col3 = (int)(gid % 768);
  const size_t bl = gid / 768;
  const int b = (int)(bl / L_DIM), l = (int)(bl % L_DIM);
  const int which = col3 >> 8, c = col3 & 255;
  const int h = c >> 5, d = c & 31;
  const int n = l >> 7, row = l & 127;

  // in-block fragment offset (independent of t)
  size_t foff;
  if (which == 0) {        // Q: A-fragment order
    const int rt = row >> 4, rlo = row & 15;
    const int lg2 = (d >> 3) & 1;
    const int e = (d & 7) + 8 * (d >> 4);
    foff = ((size_t)rt * 32 + lg2 * 16 + rlo) * 16 + e;
  } else if (which == 1) { // K^T: B-fragment order
    const int kt = row >> 4, klo = row & 15;
    foff = ((size_t)kt * 32 + ((d >> 3) & 1) * 16 + klo) * 16 + ((d & 7) + 8 * (d >> 4));
  } else {                 // V: B-fragment order
    const int kc = row >> 5, rem = row & 31;
    const int lg2 = (rem >> 3) & 1, e = (rem & 7) + 8 * (rem >> 4);
    foff = (((size_t)kc * 2 + (d >> 4)) * 32 + lg2 * 16 + (d & 15)) * 16 + e;
  }
  _Float16* dst = (which == 0) ? qs : (which == 1) ? ks : vs;

  float v = 0.f;
  for (int t = 0; t < T_DIM; ++t) {
    const float xv = qkv[((size_t)t * (B_DIM * L_DIM) + bl) * 768 + col3];
    v = 0.5f * (v + xv);                      // charge: v += (x-v)/tau, tau=2
    const float s = (v >= 1.f) ? 1.f : 0.f;   // fire
    const size_t blkbase =
        (size_t)(((t * B_DIM + b) * NWIN + n) * NH + h) * (WIN * HD);
    dst[blkbase + foff] = (_Float16)s;
    v = (s > 0.f) ? 0.f : v;                  // hard reset (detached)
  }
}

// ---------------- Flash-style routed window attention ----------------
// block = (t,b,n,h), 8 waves x 16 query rows. K/V fragments come straight from
// global (24 MB spike set is L2-resident on 192MB L2; avg 4x reuse via routing).
// Only LDS use: 16x32 P transpose buffer per wave (C-layout -> A-fragment).
__global__ __launch_bounds__(256)
void attn_kernel(const _Float16* __restrict__ qsw, const _Float16* __restrict__ ksw,
                 const _Float16* __restrict__ vsw, const int* __restrict__ idx,
                 float* __restrict__ out) {
  __shared__ __align__(16) _Float16 Ps[8][16][40];
  const int blk = blockIdx.x;               // ((t*B+b)*NWIN+n)*NH+h
  const int h = blk & 7;
  const int n = (blk >> 3) & 7;
  const int b = (blk >> 6) & 3;
  const int t = blk >> 8;
  const int tid = threadIdx.x;
  const int wave = tid >> 5, lane = tid & 31;
  const int lg = lane >> 4, lr = lane & 15;

  // Q fragment for this wave's 16 rows: one contiguous 32B chunk per lane
  const v16h aq = *(const v16h*)(
      qsw + (size_t)blk * 4096 + ((size_t)wave * 32 + lane) * 16);

  float m[8], lsum[8];
  #pragma unroll
  for (int r = 0; r < 8; ++r) { m[r] = -INFINITY; lsum[r] = 0.f; }
  v8f acc0 = {}, acc1 = {};
  const float scale = 0.17677669529663688f;   // 32^-0.5

  for (int j = 0; j < TOPK; ++j) {
    const int w = idx[((size_t)(b * NWIN + n)) * TOPK + j];
    const _Float16* kb =
        ksw + (size_t)(((t * B_DIM + b) * NWIN + w) * NH + h) * 4096;
    const _Float16* vb =
        vsw + (size_t)(((t * B_DIM + b) * NWIN + w) * NH + h) * 4096;
    #pragma unroll
    for (int cch = 0; cch < 4; ++cch) {
      const v16h bk0 = *(const v16h*)(kb + ((size_t)(cch * 2 + 0) * 32 + lane) * 16);
      const v16h bk1 = *(const v16h*)(kb + ((size_t)(cch * 2 + 1) * 32 + lane) * 16);
      // V fragments loaded early: latency hidden under score WMMAs + softmax
      const v16h bv0 = *(const v16h*)(vb + (((size_t)cch * 2 + 0) * 32 + lane) * 16);
      const v16h bv1 = *(const v16h*)(vb + (((size_t)cch * 2 + 1) * 32 + lane) * 16);
      v8f z = {};
      const v8f s0 = wmma16x16x32(aq, bk0, z);
      const v8f s1 = wmma16x16x32(aq, bk1, z);
      // online softmax over these 32 keys (row is spread over 16 lanes)
      #pragma unroll
      for (int r = 0; r < 8; ++r) {
        const float x0 = s0[r] * scale, x1 = s1[r] * scale;
        float cm = fmaxf(x0, x1);
        #pragma unroll
        for (int off = 1; off < 16; off <<= 1)
          cm = fmaxf(cm, __shfl_xor(cm, off, 32));
        const float nm = fmaxf(m[r], cm);
        const float f = __expf(m[r] - nm);
        const float p0 = __expf(x0 - nm), p1 = __expf(x1 - nm);
        float rs = p0 + p1;
        #pragma unroll
        for (int off = 1; off < 16; off <<= 1)
          rs += __shfl_xor(rs, off, 32);
        lsum[r] = lsum[r] * f + rs;
        m[r] = nm;
        acc0[r] *= f; acc1[r] *= f;
        Ps[wave][r + 8 * lg][lr]      = (_Float16)p0;   // C-layout -> LDS
        Ps[wave][r + 8 * lg][lr + 16] = (_Float16)p1;
      }
      asm volatile("s_wait_dscnt 0" ::: "memory");      // wave-local LDS RAW
      // P as A-fragment: two aligned 16B runs -> 2x ds_load_b128
      const _Float16* pp = &Ps[wave][lr][8 * lg];
      const v8h plo = *(const v8h*)pp;
      const v8h phi = *(const v8h*)(pp + 16);
      const v16h pa = __builtin_shufflevector(plo, phi,
          0,1,2,3,4,5,6,7,8,9,10,11,12,13,14,15);
      acc0 = wmma16x16x32(pa, bv0, acc0);
      acc1 = wmma16x16x32(pa, bv1, acc1);
    }
  }
  // epilogue: divide by softmax denom, write [T,B,L,C] f32
  #pragma unroll
  for (int r = 0; r < 8; ++r) {
    const float inv = 1.f / lsum[r];
    const int qrow = wave * 16 + r + 8 * lg;
    const size_t o =
        (((size_t)(t * B_DIM + b) * L_DIM) + n * WIN + qrow) * C_DIM + h * HD;
    out[o + lr]      = acc0[r] * inv;
    out[o + 16 + lr] = acc1[r] * inv;
  }
}

// ---------------- Final LIF -> binary f32 output ----------------
__global__ __launch_bounds__(256)
void lif_out_kernel(const float* __restrict__ proj, float* __restrict__ outp) {
  const size_t gid = (size_t)blockIdx.x * 256 + threadIdx.x;  // over B*L*C
  const size_t stride = (size_t)B_DIM * L_DIM * C_DIM;
  float v = 0.f;
  for (int t = 0; t < T_DIM; ++t) {
    const float xv = proj[(size_t)t * stride + gid];
    v = 0.5f * (v + xv);
    const float s = (v >= 1.f) ? 1.f : 0.f;
    outp[(size_t)t * stride + gid] = s;
    v = (s > 0.f) ? 0.f : v;
  }
}

// ---------------- launch ----------------
extern "C" void kernel_launch(void* const* d_in, const int* in_sizes, int n_in,
                              void* d_out, int out_size, void* d_ws, size_t ws_size,
                              hipStream_t stream) {
  const float* x      = (const float*)d_in[0];
  const float* w_qkv  = (const float*)d_in[1];
  const float* b_qkv  = (const float*)d_in[2];
  const float* w_proj = (const float*)d_in[3];
  const float* b_proj = (const float*)d_in[4];
  float* outp = (float*)d_out;
  char* ws = (char*)d_ws;

  // workspace layout
  int*      idx    = (int*)(ws + 0);                         // 512 B
  float*    region = (float*)(ws + 1024);                    // 32 KB
  _Float16* wswq   = (_Float16*)(ws + 65536);                // 384 KB
  _Float16* wswp   = (_Float16*)(ws + 65536 + 393216);       // 128 KB
  float*    qkv    = (float*)(ws + (1u << 20));              // 48 MB
  const size_t SZ_QKV = (size_t)T_DIM * B_DIM * L_DIM * 768 * 4;
  const size_t SZ_SPK = (size_t)T_DIM * B_DIM * L_DIM * C_DIM * 2;  // 8 MB f16
  _Float16* qs  = (_Float16*)((char*)qkv + SZ_QKV);
  _Float16* ksp = (_Float16*)((char*)qs + SZ_SPK);
  _Float16* vsp = (_Float16*)((char*)ksp + SZ_SPK);
  // qkv buffer dead after lif_qkv: recycle for attention out + proj preact
  float* attn_out = qkv;
  float* proj = (float*)((char*)qkv + (size_t)T_DIM * B_DIM * L_DIM * C_DIM * 4);

  const int M = T_DIM * B_DIM * L_DIM;   // 16384

  region_sum_kernel<<<B_DIM * NWIN, 256, 0, stream>>>(x, region);
  route_topk_kernel<<<B_DIM, 64, 0, stream>>>(region, idx);
  pack_w_kernel<<<(C_DIM * 768 + 255) / 256, 256, 0, stream>>>(w_qkv, wswq, C_DIM, 768);
  pack_w_kernel<<<(C_DIM * C_DIM + 255) / 256, 256, 0, stream>>>(w_proj, wswp, C_DIM, C_DIM);
  gemm_f16_wmma<<<dim3(768 / 128, M / 128), 256, 0, stream>>>(
      x, wswq, b_qkv, qkv, M, 768, C_DIM);
  lif_qkv_kernel<<<(B_DIM * L_DIM * 768) / 256, 256, 0, stream>>>(qkv, qs, ksp, vsp);
  attn_kernel<<<T_DIM * B_DIM * NWIN * NH, 256, 0, stream>>>(qs, ksp, vsp, idx, attn_out);
  gemm_f16_wmma<<<dim3(C_DIM / 128, M / 128), 256, 0, stream>>>(
      attn_out, wswp, b_proj, proj, M, C_DIM, C_DIM);
  lif_out_kernel<<<(B_DIM * L_DIM * C_DIM) / 256, 256, 0, stream>>>(proj, outp);
}
